// MoDESubNet2Conv_79182017069359
// MI455X (gfx1250) — matrix operand order repaired and checked by reading, not compiled, not run on hardware
//
#include <hip/hip_runtime.h>
#include <hip/hip_bf16.h>

typedef __attribute__((ext_vector_type(16))) __bf16 v16bf;
typedef __attribute__((ext_vector_type(8)))  float  v8f;

#define NS   8
#define CO   64
#define HH   256
#define WW   256
#define HP   (HH + 4)     // padded rows (2 halo each side)
#define WP   (WW + 4)     // padded cols
#define TT   8

#if __has_builtin(__builtin_amdgcn_global_load_async_to_lds_b128)
#define HAVE_ASYNC 1
#else
#define HAVE_ASYNC 0
#endif

// exact pointee type from the builtin's signature: vector_size(16) int
typedef __attribute__((__vector_size__(4 * sizeof(int)))) int vsi4;
typedef __attribute__((address_space(1))) vsi4 g_vsi4;   // global
typedef __attribute__((address_space(3))) vsi4 l_vsi4;   // LDS

// ---------------------------------------------------------------------------
// 1) Gate: g[n,e,o] = softmax_e( t[n,:] @ gw[e*Co+o,:] + gb[e*Co+o] )
// ---------------------------------------------------------------------------
__global__ void gate_kernel(const float* __restrict__ t,
                            const float* __restrict__ gw,
                            const float* __restrict__ gb,
                            float* __restrict__ g) {
  int id = blockIdx.x * blockDim.x + threadIdx.x;   // n*64 + o
  if (id >= NS * CO) return;
  int n = id >> 6, o = id & 63;
  float tv[TT];
#pragma unroll
  for (int k = 0; k < TT; ++k) tv[k] = t[n * TT + k];
  float lg[5];
  float mx = -1e30f;
#pragma unroll
  for (int e = 0; e < 5; ++e) {
    const float* row = gw + (e * CO + o) * TT;
    float s = gb[e * CO + o];
#pragma unroll
    for (int k = 0; k < TT; ++k) s += tv[k] * row[k];
    lg[e] = s;
    mx = fmaxf(mx, s);
  }
  float den = 0.f;
#pragma unroll
  for (int e = 0; e < 5; ++e) { lg[e] = __expf(lg[e] - mx); den += lg[e]; }
  float inv = 1.f / den;
#pragma unroll
  for (int e = 0; e < 5; ++e) g[(n * 5 + e) * CO + o] = lg[e] * inv;
}

// ---------------------------------------------------------------------------
// 2) Build per-sample mixed 5x5 kernels, stored as bf16 WMMA A-fragments.
// K ordering: chunk = tap*nCiC + cih ; within chunk K(0..31) follows the
// contiguous k order per lane half (same order B reads from LDS):
//   lane L: M = L&15, khalf = L>>4 ; element e(0..15):
//   K = khalf*16 + e
// flat index == global thread id: (((n*nChunks+chunk)*4+mb)*32+lane)*16+e
// ---------------------------------------------------------------------------
__global__ void build_wfrag_kernel(const float* __restrict__ g,
                                   const float* __restrict__ w5,
                                   const float* __restrict__ w3,
                                   const float* __restrict__ w1,
                                   const float* __restrict__ a3,
                                   const float* __restrict__ a5,
                                   __bf16* __restrict__ wf,
                                   int Ci, int nChunks, int nCiC) {
  int idx = blockIdx.x * blockDim.x + threadIdx.x;
  int total = NS * nChunks * 4 * 32 * 16;
  if (idx >= total) return;

  int e     = idx & 15;
  int lane  = (idx >> 4) & 31;
  int mb    = (idx >> 9) & 3;
  int ncnk  = idx >> 11;                 // n*nChunks + chunk
  int chunk = ncnk % nChunks;
  int n     = ncnk / nChunks;

  int khalf = lane >> 4;
  int K     = khalf * 16 + e;
  int tap   = chunk / nCiC;
  int cih   = chunk - tap * nCiC;
  int ci    = cih * 32 + K;
  int kh    = tap / 5, kw = tap - kh * 5;
  int o     = mb * 16 + (lane & 15);

  const float* gn = g + n * 5 * CO;
  float g0 = gn[0 * CO + o], g1 = gn[1 * CO + o], g2 = gn[2 * CO + o];
  float g3 = gn[3 * CO + o], g4 = gn[4 * CO + o];

  float val = g0 * w5[((o * Ci + ci) * 5 + kh) * 5 + kw];
  bool in3 = (kh >= 1) & (kh <= 3) & (kw >= 1) & (kw <= 3);
  if (in3) {
    val += g1 * w3[((o * Ci + ci) * 3 + (kh - 1)) * 3 + (kw - 1)];
    val += g3 * a3[o * Ci + ci] * (1.f / 27.f);
  }
  if (kh == 2 && kw == 2) val += g2 * w1[o * Ci + ci];
  val += g4 * a5[o * Ci + ci] * (1.f / 125.f);

  wf[idx] = (__bf16)val;
}

// ---------------------------------------------------------------------------
// 3) Pack NCHW f32 -> zero-padded channels-last bf16: out[n][h+2][w+2][C].
// Optionally fuses BatchNorm (biased batch stats) + ReLU (for layer-1 h).
// Tiled through LDS so both global reads and writes are coalesced.
// ---------------------------------------------------------------------------
template <int C, bool BN>
__global__ __launch_bounds__(256)
void pack_bf16_kernel(const float* __restrict__ in,
                      __bf16* __restrict__ outp,
                      const float* __restrict__ ssum,
                      const float* __restrict__ ssq,
                      const float* __restrict__ gamma,
                      const float* __restrict__ beta) {
  constexpr int TC = 2048 / C;            // cols per tile (64 or 32)
  __shared__ __bf16 tile[TC][C + 8];
  const int w0  = blockIdx.x * TC;
  const int row = blockIdx.y;
  const int n   = blockIdx.z;
  const int tid = threadIdx.x;

#pragma unroll
  for (int i = 0; i < (C * TC) / 256; ++i) {
    int e   = i * 256 + tid;
    int c   = e / TC;
    int col = e - c * TC;
    float v = in[(((size_t)n * C + c) * HH + row) * WW + w0 + col];
    if (BN) {
      const float cnt = (float)(NS * HH * WW);
      float mean = ssum[c] / cnt;
      float var  = ssq[c] / cnt - mean * mean;
      float sc   = gamma[c] * rsqrtf(var + 1e-5f);
      float bi   = beta[c] - mean * sc;
      v = fmaxf(v * sc + bi, 0.f);
    }
    tile[col][c] = (__bf16)v;
  }
  __syncthreads();
#pragma unroll
  for (int i = 0; i < (C * TC) / 256; ++i) {
    int e   = i * 256 + tid;
    int col = e / C;
    int c   = e - col * C;
    outp[(((size_t)n * HP + row + 2) * WP + w0 + col + 2) * C + c] = tile[col][c];
  }
}

// ---------------------------------------------------------------------------
// 4) Implicit-GEMM conv via v_wmma_f32_16x16x32_bf16.
// Block: 256 threads (8 waves). Tile: (n, row, 128-px strip) x 64 channels.
// wave w: mb = w&3 (16-chan block), ph = w>>2 (64-px half); 4 D-tiles/wave.
// Input is padded channels-last bf16, so LDS staging is a pure byte copy:
// 4x async global_load_async_to_lds_b128 per (row,col) position, one
// s_wait_asynccnt 0 + barrier before the K loop. LDS: [5][132][ci pad 40].
// B fragment (32x16 bf16): 32 contiguous bytes -> 2x ds_load_b128.
// Also accumulates per-channel sum / sumsq for BatchNorm.
// ---------------------------------------------------------------------------
template <int CI>
__global__ __launch_bounds__(256)
void conv_wmma_kernel(const __bf16* __restrict__ xp,   // [n][HP][WP][CI] bf16
                      const __bf16* __restrict__ wf,
                      float* __restrict__ yout,
                      float* __restrict__ ssum,
                      float* __restrict__ ssq) {
  constexpr int nCiC    = CI / 32;
  constexpr int nChunks = 25 * nCiC;
  constexpr int COLS    = 132;      // 128-px strip + 4 halo
  constexpr int CIP     = 40;       // padded ci stride (80B) -> conflict-free b128
  __shared__ __bf16 smem[5 * COLS * CIP];
  __shared__ float  sstat[2 * CO];

  const int strip = blockIdx.x;          // 0..1
  const int row   = blockIdx.y;          // 0..255
  const int n     = blockIdx.z;          // 0..7
  const int x0    = strip * 128;
  const int tid   = threadIdx.x;
  const int wave  = tid >> 5, lane = tid & 31;
  const int mb    = wave & 3,  ph   = wave >> 2;
  const int khalf = lane >> 4, pcol = lane & 15;

  if (tid < 2 * CO) sstat[tid] = 0.f;

  v8f acc[4];
#pragma unroll
  for (int t2 = 0; t2 < 4; ++t2)
#pragma unroll
    for (int r = 0; r < 8; ++r) acc[t2][r] = 0.f;

  for (int cih = 0; cih < nCiC; ++cih) {
    __syncthreads();
    // ---- stage 32 channels x 5 rows x 132 cols: async copy, 16B per lane ----
    const __bf16* gbase =
        xp + (((size_t)n * HP + row) * WP + x0) * CI + cih * 32;
    for (int idx = tid; idx < 5 * COLS * 4; idx += 256) {
      int q   = idx & 3;                  // 16B quarter of the 64B ci-run
      int pos = idx >> 2;
      int r   = pos / COLS;
      int col = pos - r * COLS;
      const __bf16* gp = gbase + ((size_t)r * WP + col) * CI + q * 8;
      __bf16* lp = &smem[(r * COLS + col) * CIP + q * 8];
#if HAVE_ASYNC
      __builtin_amdgcn_global_load_async_to_lds_b128(
          (g_vsi4*)(uintptr_t)gp,
          (l_vsi4*)(unsigned)(uintptr_t)lp, 0, 0);
#else
      *(uint4*)lp = *(const uint4*)gp;
#endif
    }
#if HAVE_ASYNC
#if __has_builtin(__builtin_amdgcn_s_wait_asynccnt)
    __builtin_amdgcn_s_wait_asynccnt(0);
#else
    asm volatile("s_wait_asynccnt 0" ::: "memory");
#endif
#endif
    __syncthreads();

    // ---- K loop: one 32-deep chunk per 5x5 tap ----
    for (int tap = 0; tap < 25; ++tap) {
      const int chunk = tap * nCiC + cih;
      const v16bf* ap = (const v16bf*)(wf +
          ((((size_t)n * nChunks + chunk) * 4 + mb) * 32 + lane) * 16);
      v16bf a = *ap;                               // 2x global_load_b128
      const int kh = tap / 5, kw = tap - kh * 5;
#pragma unroll
      for (int tile = 0; tile < 4; ++tile) {
        const int px = ph * 64 + tile * 16 + pcol;
        const int cl = px + kw;                    // col in halo'd strip
        const v16bf* bp =
            (const v16bf*)&smem[(kh * COLS + cl) * CIP + khalf * 16];
        v16bf b = *bp;                             // 2x ds_load_b128
        acc[tile] = __builtin_amdgcn_wmma_f32_16x16x32_bf16(
            false, a, false, b, (short)0, acc[tile], false, false);
      }
    }
  }

  // ---- epilogue: store raw y, accumulate BN statistics ----
#pragma unroll
  for (int r = 0; r < 8; ++r) {
    const int o = mb * 16 + khalf * 8 + r;         // 16x16 f32 D layout
    float s = 0.f, s2 = 0.f;
#pragma unroll
    for (int tile = 0; tile < 4; ++tile) {
      float v = acc[tile][r];
      s += v;
      s2 += v * v;
      const int px = ph * 64 + tile * 16 + pcol;
      yout[(((size_t)n * CO + o) * HH + row) * WW + x0 + px] = v;
    }
#pragma unroll
    for (int m = 1; m < 16; m <<= 1) {             // reduce 16-lane half
      s  += __shfl_xor(s, m, 32);
      s2 += __shfl_xor(s2, m, 32);
    }
    if (pcol == 0) {
      atomicAdd(&sstat[o], s);
      atomicAdd(&sstat[CO + o], s2);
    }
  }
  __syncthreads();
  if (tid < CO)            atomicAdd(&ssum[tid], sstat[tid]);
  else if (tid < 2 * CO)   atomicAdd(&ssq[tid - CO], sstat[tid]);
}

// ---------------------------------------------------------------------------
// 5) Final BatchNorm(training stats, biased var) + ReLU, in place, float4.
// ---------------------------------------------------------------------------
__global__ void bn_relu_kernel(float* __restrict__ buf,
                               const float* __restrict__ ssum,
                               const float* __restrict__ ssq,
                               const float* __restrict__ gamma,
                               const float* __restrict__ beta) {
  const size_t i4 = (size_t)blockIdx.x * blockDim.x + threadIdx.x;
  const size_t i  = i4 * 4;                        // [n][c][h][w], HW = 65536
  const int c = (int)((i >> 16) & 63);
  const float cnt  = (float)(NS * HH * WW);
  const float mean = ssum[c] / cnt;
  const float var  = ssq[c] / cnt - mean * mean;
  const float sc   = gamma[c] * rsqrtf(var + 1e-5f);
  const float bi   = beta[c] - mean * sc;
  float4 v = *(const float4*)(buf + i);
  v.x = fmaxf(v.x * sc + bi, 0.f);
  v.y = fmaxf(v.y * sc + bi, 0.f);
  v.z = fmaxf(v.z * sc + bi, 0.f);
  v.w = fmaxf(v.w * sc + bi, 0.f);
  *(float4*)(buf + i) = v;
}

// ---------------------------------------------------------------------------
// Launch. d_out doubles as scratch for the raw conv outputs (y1, then y2).
// ---------------------------------------------------------------------------
extern "C" void kernel_launch(void* const* d_in, const int* in_sizes, int n_in,
                              void* d_out, int out_size, void* d_ws, size_t ws_size,
                              hipStream_t stream) {
  const float* x     = (const float*)d_in[0];
  const float* t     = (const float*)d_in[1];
  const float* w5_1  = (const float*)d_in[2];
  const float* w3_1  = (const float*)d_in[3];
  const float* w1_1  = (const float*)d_in[4];
  const float* a3_1  = (const float*)d_in[5];
  const float* a5_1  = (const float*)d_in[6];
  const float* gw_1  = (const float*)d_in[7];
  const float* gb_1  = (const float*)d_in[8];
  const float* gm_1  = (const float*)d_in[9];
  const float* bt_1  = (const float*)d_in[10];
  const float* w5_2  = (const float*)d_in[11];
  const float* w3_2  = (const float*)d_in[12];
  const float* w1_2  = (const float*)d_in[13];
  const float* a3_2  = (const float*)d_in[14];
  const float* a5_2  = (const float*)d_in[15];
  const float* gw_2  = (const float*)d_in[16];
  const float* gb_2  = (const float*)d_in[17];
  const float* gm_2  = (const float*)d_in[18];
  const float* bt_2  = (const float*)d_in[19];
  float* out = (float*)d_out;

  // workspace layout (bytes)
  const size_t szXB = (size_t)NS * HP * WP * 32 * 2;   // 34,611,200
  const size_t szHB = (size_t)NS * HP * WP * 64 * 2;   // 69,222,400
  const size_t oXB  = 0;
  const size_t oHB  = oXB + szXB;
  const size_t oG1  = oHB + szHB;
  const size_t oG2  = oG1 + 16384;
  const size_t oWF1 = oG2 + 16384;                     // 819,200 used
  const size_t oWF2 = oWF1 + 1048576;                  // 1,638,400 used
  const size_t oS1  = oWF2 + 2097152;                  // sum[64]+sq[64]
  const size_t oS2  = oS1 + 1024;
  const size_t need = oS2 + 1024;
  if (ws_size < need) return;

  char* ws = (char*)d_ws;
  __bf16* xb   = (__bf16*)(ws + oXB);
  __bf16* hb   = (__bf16*)(ws + oHB);
  float*  g1   = (float*)(ws + oG1);
  float*  g2   = (float*)(ws + oG2);
  __bf16* wf1  = (__bf16*)(ws + oWF1);
  __bf16* wf2  = (__bf16*)(ws + oWF2);
  float*  sum1 = (float*)(ws + oS1);
  float*  sq1  = sum1 + CO;
  float*  sum2 = (float*)(ws + oS2);
  float*  sq2  = sum2 + CO;

  // zero padded borders (whole buffers) + stats accumulators
  (void)hipMemsetAsync(xb, 0, szXB, stream);
  (void)hipMemsetAsync(hb, 0, szHB, stream);
  (void)hipMemsetAsync(ws + oS1, 0, 2048, stream);

  // gates
  gate_kernel<<<2, 256, 0, stream>>>(t, gw_1, gb_1, g1);
  gate_kernel<<<2, 256, 0, stream>>>(t, gw_2, gb_2, g2);

  // fragment weights: layer1 Ci=32 (25 chunks), layer2 Ci=64 (50 chunks)
  build_wfrag_kernel<<<(NS * 25 * 4 * 32 * 16 + 255) / 256, 256, 0, stream>>>(
      g1, w5_1, w3_1, w1_1, a3_1, a5_1, wf1, 32, 25, 1);
  build_wfrag_kernel<<<(NS * 50 * 4 * 32 * 16 + 255) / 256, 256, 0, stream>>>(
      g2, w5_2, w3_2, w1_2, a3_2, a5_2, wf2, 64, 50, 2);

  // pack x -> padded channels-last bf16
  {
    dim3 pgrid(WW / 64, HH, NS);
    pack_bf16_kernel<32, false><<<pgrid, 256, 0, stream>>>(
        x, xb, nullptr, nullptr, nullptr, nullptr);
  }

  dim3 cgrid(2, HH, NS);
  // layer 1: xb -> d_out (raw y1) + stats
  conv_wmma_kernel<32><<<cgrid, 256, 0, stream>>>(xb, wf1, out, sum1, sq1);
  // BN+ReLU fused into pack: d_out (raw y1) -> hb (padded bf16)
  {
    dim3 pgrid(WW / 32, HH, NS);
    pack_bf16_kernel<64, true><<<pgrid, 256, 0, stream>>>(
        out, hb, sum1, sq1, gm_1, bt_1);
  }
  // layer 2: hb -> d_out (raw y2) + stats
  conv_wmma_kernel<64><<<cgrid, 256, 0, stream>>>(hb, wf2, out, sum2, sq2);
  // final BN+ReLU in place on d_out
  {
    size_t vec = (size_t)NS * CO * HH * WW / 4;
    bn_relu_kernel<<<(unsigned)(vec / 256), 256, 0, stream>>>(out, sum2, sq2, gm_2, bt_2);
  }
}
